// Net_86921548136899
// MI455X (gfx1250) — matrix operand order repaired and checked by reading, not compiled, not run on hardware
//
#include <hip/hip_runtime.h>
#include <cstdint>
#include <cstddef>

// ---------- types ----------
typedef __attribute__((ext_vector_type(16))) __bf16 v16bf;
typedef __attribute__((ext_vector_type(8)))  __bf16 v8bf;
typedef __attribute__((ext_vector_type(8)))  float  v8f;

#define DEV static __device__ __forceinline__

// ---------- wave32 reductions ----------
DEV float wave_max32(float v) {
#pragma unroll
  for (int off = 16; off; off >>= 1) v = fmaxf(v, __shfl_xor(v, off, 32));
  return v;
}
DEV float wave_sum32(float v) {
#pragma unroll
  for (int off = 16; off; off >>= 1) v += __shfl_xor(v, off, 32);
  return v;
}

// ---------- MX block quantization (BLOCK=32, MBITS=7) ----------
// scale = 2^(floor(log2(maxabs or 1)) - 6); q = clip(rint(x/scale), -127, 127) * scale
DEV float mx_quant(float x, float maxabs) {
  float m = maxabs > 0.f ? maxabs : 1.f;
  int e;
  (void)frexpf(m, &e);                 // m = f * 2^e, f in [0.5,1) -> floor(log2 m) = e-1
  float q = rintf(x * ldexpf(1.f, 7 - e));
  q = fminf(127.f, fmaxf(-127.f, q));
  return q * ldexpf(1.f, e - 7);
}

// ---------- bf16 helpers (storage = unsigned short; exact for MX values) ----------
DEV unsigned short f2bf(float f) {
  unsigned u = __float_as_uint(f);
  u += 0x7FFFu + ((u >> 16) & 1u);     // RNE
  return (unsigned short)(u >> 16);
}
DEV float bf2f(unsigned short h) { return __uint_as_float(((unsigned)h) << 16); }

// ---------- problem constants ----------
static constexpr int    BATCH = 4096;
// workspace layout (bytes), all 256B-aligned by construction
static constexpr size_t OFF_H1Q  = 0;                                          // bf16 [B*676*32]  conv1 out, quantized, NHWC
static constexpr size_t OFF_H2   = OFF_H1Q  + (size_t)BATCH * 676 * 32 * 2;    // bf16 [B*576*64]  conv2 out (relu), NHWC
static constexpr size_t OFF_A1Q  = OFF_H2   + (size_t)BATCH * 576 * 64 * 2;    // bf16 [B*9216]    pooled+flattened, quantized
static constexpr size_t OFF_H3   = OFF_A1Q  + (size_t)BATCH * 9216 * 2;        // f32  [B*128]     fc1 out (relu)
static constexpr size_t OFF_WQ2  = OFF_H3   + (size_t)BATCH * 128 * 4;         // bf16 [9*64*32]   conv2 W, WMMA-B packed
static constexpr size_t OFF_W1Q  = OFF_WQ2  + (size_t)9 * 64 * 32 * 2;         // bf16 [128*9216]  fc1 W, quantized
static constexpr size_t OFF_W2QF = OFF_W1Q  + (size_t)128 * 9216 * 2;          // f32  [10*128]    fc2 W, quantized
static constexpr size_t OFF_W1QF = OFF_W2QF + (size_t)10 * 128 * 4;            // f32  [32*9]      conv1 W, quantized

// =======================================================================
// Weight pack kernels (MX quantization along the contraction dimension)
// =======================================================================
__global__ void k_pack_conv1w(const float* __restrict__ w, float* __restrict__ wq) {
  // w: [32,1,3,3]; axis=1 has size 1 -> per-element quantization
  int i = threadIdx.x;                    // 0..287
  float v = w[i];
  wq[i] = mx_quant(v, fabsf(v));
}

__global__ void k_pack_conv2w(const float* __restrict__ w, unsigned short* __restrict__ wq) {
  // w: [O=64][I=32][3][3]; block over I=32 per (o,tap). Pack as [tap][n=o][k=i].
  int t = blockIdx.x, n = blockIdx.y, k = threadIdx.x;  // t in 0..8, n in 0..63, k in 0..31
  int dy = t / 3, dx = t % 3;
  float v = w[((size_t)(n * 32 + k) * 3 + dy) * 3 + dx];
  float m = wave_max32(fabsf(v));
  wq[((size_t)t * 64 + n) * 32 + k] = f2bf(mx_quant(v, m));
}

__global__ void k_pack_fc1w(const float* __restrict__ w, unsigned short* __restrict__ wq) {
  // w: [128][9216]; blocks of 32 along "in"
  int blk = blockIdx.x, n = blockIdx.y, k = threadIdx.x;
  size_t idx = (size_t)n * 9216 + blk * 32 + k;
  float v = w[idx];
  float m = wave_max32(fabsf(v));
  wq[idx] = f2bf(mx_quant(v, m));
}

__global__ void k_pack_fc2w(const float* __restrict__ w, float* __restrict__ wq) {
  // w: [10][128]; blocks of 32 along "in"; keep f32 (exact)
  int o = blockIdx.x >> 2, blk = blockIdx.x & 3, k = threadIdx.x;
  size_t idx = (size_t)o * 128 + blk * 32 + k;
  float v = w[idx];
  float m = wave_max32(fabsf(v));
  wq[idx] = mx_quant(v, m);
}

// =======================================================================
// conv1: 1->32 channels, 3x3 VALID. One wave per output pixel; lane = channel.
// Per-element MX quant of inputs (C=1 blocks), then channel-block (32 = wave)
// quantization of the relu'd output fused into the epilogue. NHWC bf16 out.
// =======================================================================
__global__ void __launch_bounds__(256) k_conv1(const float* __restrict__ x,
                                               const float* __restrict__ wq,
                                               const float* __restrict__ bias,
                                               unsigned short* __restrict__ h1q) {
  const int lane = threadIdx.x & 31;
  const int p    = blockIdx.x * 8 + (threadIdx.x >> 5);  // pixel id, exact cover
  const int b  = p / 676, s = p % 676;
  const int oy = s / 26,  ox = s % 26;
  const float* xp = x + (size_t)b * 784;
  float acc = bias[lane];
#pragma unroll
  for (int dy = 0; dy < 3; ++dy)
#pragma unroll
    for (int dx = 0; dx < 3; ++dx) {
      float xv = xp[(oy + dy) * 28 + (ox + dx)];
      float xq = mx_quant(xv, fabsf(xv));
      acc += xq * wq[lane * 9 + dy * 3 + dx];
    }
  acc = fmaxf(acc, 0.f);
  float bm = wave_max32(fabsf(acc));                     // block = 32 channels = wave
  h1q[(size_t)p * 32 + lane] = f2bf(mx_quant(acc, bm));
}

// =======================================================================
// conv2 as implicit GEMM with bf16 WMMA:
//   M = B*24*24 pixels, N = 64 out-channels, K = 9 taps * 32 channels.
// Each wave owns a 16(M) x 64(N) strip: 4 f32 accumulators, 9 taps x 4 WMMAs.
// Weights (36 KB, pre-packed in B-operand order) staged once per WG in LDS.
// Epilogue: bias + relu + bf16 store (NHWC).
// =======================================================================
__global__ void __launch_bounds__(256) k_conv2(const unsigned short* __restrict__ h1q,
                                               const unsigned short* __restrict__ wq2,
                                               const float* __restrict__ bias,
                                               unsigned short* __restrict__ h2) {
  __shared__ alignas(16) unsigned short sw[9 * 64 * 32];  // 36,864 B
  for (int i = threadIdx.x; i < (9 * 64 * 32) / 8; i += 256)
    ((uint4*)sw)[i] = ((const uint4*)wq2)[i];
  __syncthreads();

  const int lane = threadIdx.x & 31;
  const int nloc = lane & 15;           // N (B/C col) or A row
  const int half = lane >> 4;           // lane half selects K-chunk / M+8
  const int wid    = blockIdx.x * 8 + (threadIdx.x >> 5);
  const int nWaves = gridDim.x * 8;
  const int nStrips = (BATCH * 576) / 16;  // 147,456

  for (int strip = wid; strip < nStrips; strip += nWaves) {
    const int m0 = strip * 16;
    const int p  = m0 + nloc;           // this lane's A row (pixel)
    const int b  = p / 576, s = p % 576;
    const int oy = s / 24,  ox = s % 24;

    v8f acc[4] = {v8f{}, v8f{}, v8f{}, v8f{}};
#pragma unroll
    for (int t = 0; t < 9; ++t) {
      const int dy = t / 3, dx = t % 3;
      // A (16x32 bf16): lane half 0 -> K 0..15; half 1 -> K 16..31 (contiguous)
      const unsigned short* ap =
          h1q + (((size_t)b * 26 + (oy + dy)) * 26 + (ox + dx)) * 32 + half * 16;
      v8bf alo = *reinterpret_cast<const v8bf*>(ap);
      v8bf ahi = *reinterpret_cast<const v8bf*>(ap + 8);
      v16bf A = __builtin_shufflevector(alo, ahi,
                 0,1,2,3,4,5,6,7,8,9,10,11,12,13,14,15);
      // B (32x16 bf16): lane nloc = column, half selects K 0..15 / 16..31 (contiguous)
      const unsigned short* sb = sw + (size_t)t * 64 * 32 + half * 16;
#pragma unroll
      for (int j = 0; j < 4; ++j) {
        v16bf Bj = *reinterpret_cast<const v16bf*>(sb + (j * 16 + nloc) * 32);
        acc[j] = __builtin_amdgcn_wmma_f32_16x16x32_bf16(
            false, A, false, Bj, (short)0, acc[j], false, false);
      }
    }
    // epilogue: C layout -> VGPR r, lane: M = r + 8*half, N = nloc (+16j)
#pragma unroll
    for (int j = 0; j < 4; ++j) {
      const int n  = j * 16 + nloc;
      const float bb = bias[n];
#pragma unroll
      for (int r = 0; r < 8; ++r) {
        const int pm = m0 + r + half * 8;
        float v = fmaxf(acc[j][r] + bb, 0.f);
        h2[(size_t)pm * 64 + n] = f2bf(v);
      }
    }
  }
}

// =======================================================================
// maxpool 2x2 fused with flat-dim MX quantization. One wave = one 32-wide
// block of the flattened [64*12*12]=9216 features (blocks straddle channels).
// =======================================================================
__global__ void __launch_bounds__(256) k_poolq(const unsigned short* __restrict__ h2,
                                               unsigned short* __restrict__ a1q) {
  const int lane = threadIdx.x & 31;
  const int wid  = blockIdx.x * 8 + (threadIdx.x >> 5);
  const int b = wid / 288, blk = wid % 288;
  const int f = blk * 32 + lane;        // flat index = c*144 + py*12 + px
  const int c = f / 144, s = f % 144;
  const int py = s / 12, px = s % 12;
  const size_t base = (((size_t)b * 24 + py * 2) * 24 + px * 2) * 64 + c;
  float v = bf2f(h2[base]);
  v = fmaxf(v, bf2f(h2[base + 64]));
  v = fmaxf(v, bf2f(h2[base + 24 * 64]));
  v = fmaxf(v, bf2f(h2[base + 24 * 64 + 64]));
  float m = wave_max32(fabsf(v));
  a1q[(size_t)b * 9216 + f] = f2bf(mx_quant(v, m));
}

// =======================================================================
// fc1: [4096,9216] x [9216,128] with bf16 WMMA, 288 K-steps of 32.
// One wave per 16x16 output tile; bias + relu fused.
// =======================================================================
__global__ void __launch_bounds__(256) k_fc1(const unsigned short* __restrict__ a1q,
                                             const unsigned short* __restrict__ w1q,
                                             const float* __restrict__ bias,
                                             float* __restrict__ h3) {
  const int lane = threadIdx.x & 31;
  const int nloc = lane & 15, half = lane >> 4;
  const int wid = blockIdx.x * 8 + (threadIdx.x >> 5);
  const int m0 = (wid >> 3) * 16;       // 256 M-tiles
  const int n0 = (wid & 7) * 16;        // 8 N-tiles
  const unsigned short* arow = a1q + (size_t)(m0 + nloc) * 9216 + half * 16;
  const unsigned short* brow = w1q + (size_t)(n0 + nloc) * 9216 + half * 16;
  v8f acc{};
  for (int kk = 0; kk < 9216; kk += 32) {
    v8bf alo = *reinterpret_cast<const v8bf*>(arow + kk);
    v8bf ahi = *reinterpret_cast<const v8bf*>(arow + kk + 8);
    v16bf A = __builtin_shufflevector(alo, ahi,
               0,1,2,3,4,5,6,7,8,9,10,11,12,13,14,15);
    v16bf Bv = *reinterpret_cast<const v16bf*>(brow + kk);
    acc = __builtin_amdgcn_wmma_f32_16x16x32_bf16(
        false, A, false, Bv, (short)0, acc, false, false);
  }
  const int n = n0 + nloc;
  const float bb = bias[n];
#pragma unroll
  for (int r = 0; r < 8; ++r) {
    const int m = m0 + r + half * 8;
    h3[(size_t)m * 128 + n] = fmaxf(acc[r] + bb, 0.f);
  }
}

// =======================================================================
// fc2 (K=128, N=10) + log_softmax. One wave per batch row; x is MX-quantized
// in-register (4 blocks of 32 = 4 wave reductions), weights pre-quantized.
// =======================================================================
__global__ void __launch_bounds__(256) k_fc2(const float* __restrict__ h3,
                                             const float* __restrict__ wqf,
                                             const float* __restrict__ bias,
                                             float* __restrict__ out) {
  const int lane = threadIdx.x & 31;
  const int row  = blockIdx.x * 8 + (threadIdx.x >> 5);
  const float* xr = h3 + (size_t)row * 128;
  float xq[4];
#pragma unroll
  for (int j = 0; j < 4; ++j) {
    float v = xr[j * 32 + lane];
    float m = wave_max32(fabsf(v));
    xq[j] = mx_quant(v, m);
  }
  float lg[10];
#pragma unroll
  for (int o = 0; o < 10; ++o) {
    float p = 0.f;
#pragma unroll
    for (int j = 0; j < 4; ++j) p += xq[j] * wqf[o * 128 + j * 32 + lane];
    lg[o] = wave_sum32(p) + bias[o];
  }
  float mx = lg[0];
#pragma unroll
  for (int o = 1; o < 10; ++o) mx = fmaxf(mx, lg[o]);
  float se = 0.f;
#pragma unroll
  for (int o = 0; o < 10; ++o) se += expf(lg[o] - mx);
  const float lse = mx + logf(se);
  if (lane < 10) out[(size_t)row * 10 + lane] = lg[lane] - lse;
}

// =======================================================================
extern "C" void kernel_launch(void* const* d_in, const int* in_sizes, int n_in,
                              void* d_out, int out_size, void* d_ws, size_t ws_size,
                              hipStream_t stream) {
  (void)in_sizes; (void)n_in; (void)out_size; (void)ws_size;
  const float* x   = (const float*)d_in[0];
  const float* c1w = (const float*)d_in[1];
  const float* c1b = (const float*)d_in[2];
  const float* c2w = (const float*)d_in[3];
  const float* c2b = (const float*)d_in[4];
  const float* f1w = (const float*)d_in[5];
  const float* f1b = (const float*)d_in[6];
  const float* f2w = (const float*)d_in[7];
  const float* f2b = (const float*)d_in[8];

  char* ws = (char*)d_ws;
  unsigned short* h1q  = (unsigned short*)(ws + OFF_H1Q);
  unsigned short* h2   = (unsigned short*)(ws + OFF_H2);
  unsigned short* a1q  = (unsigned short*)(ws + OFF_A1Q);
  float*          h3   = (float*)(ws + OFF_H3);
  unsigned short* wq2  = (unsigned short*)(ws + OFF_WQ2);
  unsigned short* w1q  = (unsigned short*)(ws + OFF_W1Q);
  float*          w2qf = (float*)(ws + OFF_W2QF);
  float*          w1qf = (float*)(ws + OFF_W1QF);

  // weight packing (cheap, deterministic, every call)
  k_pack_conv1w<<<1, 288, 0, stream>>>(c1w, w1qf);
  k_pack_conv2w<<<dim3(9, 64), 32, 0, stream>>>(c2w, wq2);
  k_pack_fc1w<<<dim3(288, 128), 32, 0, stream>>>(f1w, w1q);
  k_pack_fc2w<<<40, 32, 0, stream>>>(f2w, w2qf);

  // forward pass
  k_conv1<<<(BATCH * 676) / 8, 256, 0, stream>>>(x, w1qf, c1b, h1q);
  k_conv2<<<2304, 256, 0, stream>>>(h1q, wq2, c2b, h2);
  k_poolq<<<(BATCH * 288) / 8, 256, 0, stream>>>(h2, a1q);
  k_fc1<<<256, 256, 0, stream>>>(a1q, w1q, f1b, h3);
  k_fc2<<<512, 256, 0, stream>>>(h3, w2qf, f2b, (float*)d_out);
}
